// GatedESNGRU_62912680952221
// MI455X (gfx1250) — matrix operand order repaired and checked by reading, not compiled, not run on hardware
//
#include <hip/hip_runtime.h>
#include <hip/hip_bf16.h>
#include <math.h>

// ---------------------------------------------------------------------------
// GatedESNGRU for MI455X (gfx1250):
//   - bf16 WMMA (16x16x32, f32 accum) for all GEMMs
//   - TDM (tensor_load_to_lds) stages per-block A tiles into LDS with
//     bank-rotation padding; A fragments read back via ds_load_b128
//   - B panels stream from L2 (weights are L2-resident: ~12 MB bf16 vs 192 MB)
// B=64, T=512, I=128, R=2048, H=512, O=128, LEAK=0.3
// ---------------------------------------------------------------------------

typedef __attribute__((ext_vector_type(16))) __bf16  v16bf;
typedef __attribute__((ext_vector_type(8)))  float   v8f;
typedef __attribute__((ext_vector_type(4)))  unsigned int u32x4;
typedef __attribute__((ext_vector_type(8)))  int     i32x8;
typedef __attribute__((ext_vector_type(4)))  int     i32x4;

// LDS (address_space(3)) element / vector types
typedef __attribute__((address_space(3))) unsigned short lds_us;
typedef __attribute__((address_space(3))) const u32x4    lds_cu32x4;

#define LEAK_F 0.3f

static __device__ __forceinline__ unsigned short f2bf_bits(float f) {
  unsigned u = __float_as_uint(f);
  unsigned r = u + 0x7FFFu + ((u >> 16) & 1u);   // round-to-nearest-even
  return (unsigned short)(r >> 16);
}

// A fragment (16x32 bf16) from global, p points at element (row, kbase):
//   lane<16 -> K = +0..7 and +16..23 ; lane>=16 -> K = +8..15 and +24..31
static __device__ __forceinline__ v16bf load_frag_a(const unsigned short* p,
                                                    int lane) {
  int o = (lane & 16) ? 8 : 0;
  union { u32x4 v[2]; v16bf b; } u;
  u.v[0] = *(const u32x4*)(p + o);
  u.v[1] = *(const u32x4*)(p + o + 16);
  return u.b;
}

// Same fragment pattern, but from LDS (lowers to ds_load_b128).
static __device__ __forceinline__ v16bf load_frag_a_lds(const lds_us* p,
                                                        int lane) {
  int o = (lane & 16) ? 8 : 0;
  union { u32x4 v[2]; v16bf b; } u;
  u.v[0] = *(lds_cu32x4*)(p + o);
  u.v[1] = *(lds_cu32x4*)(p + o + 16);
  return u.b;
}

// B fragment from packed layout: 512-elem block, element (k,n) at (n&15)*32+k.
static __device__ __forceinline__ v16bf load_b(const unsigned short* blk,
                                               int lane) {
  const unsigned short* p = blk + ((lane & 15) << 5) + ((lane & 16) ? 8 : 0);
  union { u32x4 v[2]; v16bf b; } u;
  u.v[0] = *(const u32x4*)(p);
  u.v[1] = *(const u32x4*)(p + 16);
  return u.b;
}

static __device__ __forceinline__ v8f wmma_bf16(v16bf a, v16bf b, v8f c) {
  return __builtin_amdgcn_wmma_f32_16x16x32_bf16(false, a, false, b,
                                                 (short)0, c, false, false);
}

static __device__ __forceinline__ float sigm(float x) {
  return 1.0f / (1.0f + __expf(-x));
}

// --------------------------- TDM descriptor --------------------------------
// 2D tile load, bf16 elements (data_size=2B), LDS padding enabled:
// pad_amount = 4 DWORDs (16B, keeps ds_load_b128 alignment) every
// 2^(pad_interval+1) DWORDs, rotating rows across LDS banks.
static __device__ __forceinline__ void tdm_load_2d(
    unsigned lds_off, const void* gptr,
    unsigned td0, unsigned td1, unsigned tile0, unsigned tile1,
    unsigned long long stride0_elems,
    unsigned pad_interval /*code*/, unsigned pad_amount /*code*/) {
  unsigned long long ga = (unsigned long long)gptr;
  u32x4 g0;
  g0[0] = 1u;                                   // count=1, user descriptor
  g0[1] = lds_off;                              // LDS byte address
  g0[2] = (unsigned)ga;                         // global_addr[31:0]
  g0[3] = ((unsigned)(ga >> 32) & 0x01ffffffu)  // global_addr[56:32]
        | (2u << 30);                           // type = 2 ("image")
  i32x8 g1;
  g1[0] = (int)((1u << 16)                      // data_size = 2 bytes
              | (1u << 20)                      // pad_enable
              | (pad_interval << 22)
              | (pad_amount << 25));
  g1[1] = (int)((td0 & 0xffffu) << 16);                              // tensor_dim0 lo
  g1[2] = (int)(((td0 >> 16) & 0xffffu) | ((td1 & 0xffffu) << 16));  // d0 hi | d1 lo
  g1[3] = (int)(((td1 >> 16) & 0xffffu) | ((tile0 & 0xffffu) << 16));// d1 hi | tile0
  g1[4] = (int)(tile1 & 0xffffu);                                    // tile1, tile2=0
  g1[5] = (int)(unsigned)(stride0_elems & 0xffffffffull);            // dim0 stride lo
  g1[6] = (int)(unsigned)((stride0_elems >> 32) & 0xffffull);        // stride hi
  g1[7] = 0;
  i32x4 z4 = {0, 0, 0, 0};
  i32x8 z8 = {0, 0, 0, 0, 0, 0, 0, 0};
  // 6-arg form (clang-23 / therock-10.0 signature)
  __builtin_amdgcn_tensor_load_to_lds(g0, g1, z4, z4, z8, 0);
}

// LDS geometry after TDM padding (pad = 16B per chunk):
//   ESN h tile: rows of 2048 bf16 = 4 chunks of 1024B (+16B) -> row stride 4160B
//   GRU h tile: rows of 512 bf16  = 1 chunk of 1024B (+16B)  -> row stride 1040B
//   x tile:     rows of 128 bf16  = 1 chunk of 256B  (+16B)  -> row stride 272B
// All in bf16 elements below (half the byte counts).
#define ESN_ROW_E   2080         // 4160 B
#define ESN_CHUNK_E 520          // 1040 B
#define GRU_ROW_E   520          // 1040 B
#define X_ROW_E     136          // 272 B
#define X_LDS_OFF_B 66560        // 16 * 4160
#define X_LDS_OFF_E 33280
#define LDS_BYTES   (X_LDS_OFF_B + 16 * 272)   // 70912

// --------------------------- prep kernels ----------------------------------

__global__ void pack_b_kernel(const float* __restrict__ src,
                              unsigned short* __restrict__ dst,
                              int K, int N, int ks, int ns) {
  int idx = blockIdx.x * blockDim.x + threadIdx.x;
  if (idx >= K * N) return;
  int k = idx / N;
  int n = idx - k * N;
  float v = src[(size_t)k * ks + (size_t)n * ns];
  size_t off = (((size_t)(n >> 4) * (K >> 5) + (k >> 5)) << 9)
             + ((n & 15) << 5) + (k & 31);
  dst[off] = f2bf_bits(v);
}

__global__ void f32_to_bf16_kernel(const float* __restrict__ src,
                                   unsigned short* __restrict__ dst, int n) {
  int i = blockIdx.x * blockDim.x + threadIdx.x;
  if (i < n) dst[i] = f2bf_bits(src[i]);
}

__global__ void zero_kernel(unsigned int* __restrict__ p, int n) {
  int i = blockIdx.x * blockDim.x + threadIdx.x;
  if (i < n) p[i] = 0u;
}

// --------------------------- fused ESN+GRU step ----------------------------
// Blocks 0..63 : ESN. mt = b>>4, 8 waves cover nt = (b&15)*8 + wave (0..127).
// Blocks 64..79: GRU. mt = (b-64)>>2, nt = ((b-64)&3)*8 + wave (0..31).
// Each block TDM-stages its 16-row A tiles (h state + x_t) into LDS once.
__global__ __launch_bounds__(256) void step_kernel(
    const unsigned short* __restrict__ xbf, int t,
    const unsigned short* __restrict__ WinP,
    const unsigned short* __restrict__ WresP,
    const unsigned short* __restrict__ WihP,
    const unsigned short* __restrict__ WhhP,
    const float* __restrict__ bih, const float* __restrict__ bhh,
    const float* __restrict__ hesn_in, const unsigned short* __restrict__ hesn_in_bf,
    float* __restrict__ hesn_out, unsigned short* __restrict__ hesn_out_bf,
    const float* __restrict__ hgru_in, const unsigned short* __restrict__ hgru_in_bf,
    float* __restrict__ hgru_out, unsigned short* __restrict__ hgru_out_bf)
{
  __shared__ __align__(16) unsigned char sTile[LDS_BYTES];
  int wave = threadIdx.x >> 5;
  int lane = threadIdx.x & 31;
  int lrow = lane & 15;
  int b    = blockIdx.x;

  if (b < 64) {
    // =========================== ESN ===========================
    int mt = b >> 4;
    int nt = (b & 15) * 8 + wave;
    if (wave == 0) {
      // h tile: 16 x 2048, pad_interval=7 (256 DW), pad_amount=3 (4 DW)
      tdm_load_2d(0, hesn_in_bf + (size_t)mt * 16 * 2048,
                  2048, 16, 2048, 16, 2048ull, 7u, 3u);
      // x tile: 16 x 128 (row stride T*I), pad_interval=5 (64 DW)
      tdm_load_2d(X_LDS_OFF_B,
                  xbf + ((size_t)mt * 16 * 512 + (size_t)t) * 128,
                  128, 16, 128, 16, 65536ull, 5u, 3u);
      __builtin_amdgcn_s_wait_tensorcnt(0);
    }
    __syncthreads();
    // Launder LDS base: escapes into asm with memory clobber so the compiler
    // must treat the (TDM-written) LDS contents as defined and re-load them.
    // Pointer keeps address_space(3) so loads lower to ds_load_b128.
    lds_us* sbase = (lds_us*)sTile;
    asm volatile("" : "+v"(sbase) :: "memory");

    v8f acc = {};
#pragma unroll
    for (int kb = 0; kb < 4; ++kb) {          // X_t (64x128) @ W_in tile
      const lds_us* pa = sbase + X_LDS_OFF_E + lrow * X_ROW_E + kb * 32;
      v16bf a = load_frag_a_lds(pa, lane);
      v16bf bfr = load_b(WinP + (((size_t)nt * 4 + kb) << 9), lane);
      acc = wmma_bf16(a, bfr, acc);
    }
#pragma unroll 4
    for (int kb = 0; kb < 64; ++kb) {         // h (64x2048) @ W_res tile
      const unsigned short* blk = WresP + (((size_t)nt * 64 + kb) << 9);
      if (kb + 1 < 64)                        // gfx1250 global_prefetch_b8
        __builtin_prefetch(blk + 512, 0, 1);
      int kbase = kb * 32;
      const lds_us* pa = sbase + lrow * ESN_ROW_E
                       + (kbase >> 9) * ESN_CHUNK_E + (kbase & 511);
      v16bf a = load_frag_a_lds(pa, lane);
      v16bf bfr = load_b(blk, lane);
      acc = wmma_bf16(a, bfr, acc);
    }
    int col   = nt * 16 + lrow;
    int rbase = mt * 16 + ((lane & 16) ? 8 : 0);
#pragma unroll
    for (int k = 0; k < 8; ++k) {
      int row = rbase + k;
      float hold = hesn_in[(size_t)row * 2048 + col];
      float cand = tanhf(acc[k]);
      float hnew = (1.0f - LEAK_F) * hold + LEAK_F * cand;
      hesn_out[(size_t)row * 2048 + col]    = hnew;
      hesn_out_bf[(size_t)row * 2048 + col] = f2bf_bits(hnew);
    }
  } else {
    // =========================== GRU ===========================
    int q  = b - 64;
    int mt = q >> 2;
    int nt = (q & 3) * 8 + wave;              // 0..31
    if (wave == 0) {
      // h tile: 16 x 512, pad_interval=7, pad_amount=3
      tdm_load_2d(0, hgru_in_bf + (size_t)mt * 16 * 512,
                  512, 16, 512, 16, 512ull, 7u, 3u);
      tdm_load_2d(X_LDS_OFF_B,
                  xbf + ((size_t)mt * 16 * 512 + (size_t)t) * 128,
                  128, 16, 128, 16, 65536ull, 5u, 3u);
      __builtin_amdgcn_s_wait_tensorcnt(0);
    }
    __syncthreads();
    lds_us* sbase = (lds_us*)sTile;
    asm volatile("" : "+v"(sbase) :: "memory");

    v8f r = {}, z = {}, xn = {}, hn = {};
#pragma unroll
    for (int kb = 0; kb < 4; ++kb) {          // x-part (K=128), W_ih^T N=1536
      const lds_us* pa = sbase + X_LDS_OFF_E + lrow * X_ROW_E + kb * 32;
      v16bf a = load_frag_a_lds(pa, lane);
      r  = wmma_bf16(a, load_b(WihP + (((size_t)(nt)      * 4 + kb) << 9), lane), r);
      z  = wmma_bf16(a, load_b(WihP + (((size_t)(nt + 32) * 4 + kb) << 9), lane), z);
      xn = wmma_bf16(a, load_b(WihP + (((size_t)(nt + 64) * 4 + kb) << 9), lane), xn);
    }
#pragma unroll 4
    for (int kb = 0; kb < 16; ++kb) {         // h-part (K=512), W_hh^T N=1536
      const lds_us* pa = sbase + lrow * GRU_ROW_E + kb * 32;
      v16bf a = load_frag_a_lds(pa, lane);
      r  = wmma_bf16(a, load_b(WhhP + (((size_t)(nt)      * 16 + kb) << 9), lane), r);
      z  = wmma_bf16(a, load_b(WhhP + (((size_t)(nt + 32) * 16 + kb) << 9), lane), z);
      hn = wmma_bf16(a, load_b(WhhP + (((size_t)(nt + 64) * 16 + kb) << 9), lane), hn);
    }
    int j = nt * 16 + lrow;
    float bihr = bih[j], bihz = bih[512 + j], bihn = bih[1024 + j];
    float bhhr = bhh[j], bhhz = bhh[512 + j], bhhn = bhh[1024 + j];
    int rbase = mt * 16 + ((lane & 16) ? 8 : 0);
#pragma unroll
    for (int k = 0; k < 8; ++k) {
      int row = rbase + k;
      float hold = hgru_in[(size_t)row * 512 + j];
      float rv = sigm(r[k] + bihr + bhhr);
      float zv = sigm(z[k] + bihz + bhhz);
      float nv = tanhf(xn[k] + bihn + rv * (hn[k] + bhhn));
      float hnew = (1.0f - zv) * nv + zv * hold;
      hgru_out[(size_t)row * 512 + j]    = hnew;
      hgru_out_bf[(size_t)row * 512 + j] = f2bf_bits(hnew);
    }
  }
}

// --------------------------- tail GEMM (proj/gate/head) --------------------
//   mode 0: identity (proj: f32+bf16)   mode 1: sigmoid-gate mix (bf16)
//   mode 2: relu (bf16)                 mode 3: identity (f32 final out)
__global__ __launch_bounds__(256) void tail_gemm_kernel(
    const unsigned short* __restrict__ A1, int lda1,
    const unsigned short* __restrict__ A2, int lda2, int kb_split,
    const unsigned short* __restrict__ Bpack, int Kb, int Nt,
    const float* __restrict__ bias, int mode,
    const float* __restrict__ aux1, const float* __restrict__ aux2, int ld_aux,
    float* __restrict__ out_f, unsigned short* __restrict__ out_bf, int ld_out)
{
  int wave = threadIdx.x >> 5, lane = threadIdx.x & 31;
  int gid = blockIdx.x * 8 + wave;
  if (gid >= 4 * Nt) return;
  int mt = gid & 3, nt = gid >> 2;
  int lrow = lane & 15;
  v8f acc = {};
  for (int kb = 0; kb < Kb; ++kb) {
    const unsigned short* arow;
    int kbase;
    if (kb < kb_split) { arow = A1 + (size_t)(mt * 16 + lrow) * lda1; kbase = kb * 32; }
    else               { arow = A2 + (size_t)(mt * 16 + lrow) * lda2; kbase = (kb - kb_split) * 32; }
    v16bf a = load_frag_a(arow + kbase, lane);
    v16bf bfr = load_b(Bpack + (((size_t)nt * Kb + kb) << 9), lane);
    acc = wmma_bf16(a, bfr, acc);
  }
  int col = nt * 16 + lrow;
  float bv = bias[col];
  int rbase = mt * 16 + ((lane & 16) ? 8 : 0);
#pragma unroll
  for (int k = 0; k < 8; ++k) {
    int row = rbase + k;
    float v = acc[k] + bv;
    if (mode == 1) {
      float g  = sigm(v);
      float pv = aux1[(size_t)row * ld_aux + col];
      float gv = aux2[(size_t)row * ld_aux + col];
      v = g * pv + (1.0f - g) * gv;
    } else if (mode == 2) {
      v = v > 0.0f ? v : 0.0f;
    }
    if (out_f)  out_f[(size_t)row * ld_out + col]  = v;
    if (out_bf) out_bf[(size_t)row * ld_out + col] = f2bf_bits(v);
  }
}

// ---------------------------------------------------------------------------

extern "C" void kernel_launch(void* const* d_in, const int* in_sizes, int n_in,
                              void* d_out, int out_size, void* d_ws, size_t ws_size,
                              hipStream_t stream) {
  (void)in_sizes; (void)n_in; (void)out_size; (void)ws_size;
  const float* x       = (const float*)d_in[0];
  const float* W_in    = (const float*)d_in[1];
  const float* W_res   = (const float*)d_in[2];
  const float* gW_ih   = (const float*)d_in[3];
  const float* gW_hh   = (const float*)d_in[4];
  const float* gb_ih   = (const float*)d_in[5];
  const float* gb_hh   = (const float*)d_in[6];
  const float* proj_W  = (const float*)d_in[7];
  const float* proj_b  = (const float*)d_in[8];
  const float* gate_W  = (const float*)d_in[9];
  const float* gate_b  = (const float*)d_in[10];
  const float* head_W1 = (const float*)d_in[11];
  const float* head_b1 = (const float*)d_in[12];
  const float* head_W2 = (const float*)d_in[13];
  const float* head_b2 = (const float*)d_in[14];
  float* out = (float*)d_out;

  char* ws = (char*)d_ws;
  size_t off = 0;
  auto alloc = [&](size_t bytes) -> char* {
    char* p = ws + off;
    off = (off + bytes + 255) & ~(size_t)255;
    return p;
  };
  unsigned short* xbf   = (unsigned short*)alloc((size_t)64 * 512 * 128 * 2);
  unsigned short* WinP  = (unsigned short*)alloc((size_t)128 * 2048 * 2);
  unsigned short* WresP = (unsigned short*)alloc((size_t)2048 * 2048 * 2);
  unsigned short* WihP  = (unsigned short*)alloc((size_t)128 * 1536 * 2);
  unsigned short* WhhP  = (unsigned short*)alloc((size_t)512 * 1536 * 2);
  unsigned short* projP = (unsigned short*)alloc((size_t)2048 * 512 * 2);
  unsigned short* gateP = (unsigned short*)alloc((size_t)1024 * 512 * 2);
  unsigned short* h1P   = (unsigned short*)alloc((size_t)512 * 512 * 2);
  unsigned short* h2P   = (unsigned short*)alloc((size_t)512 * 128 * 2);
  float* hesn_f[2];  unsigned short* hesn_b[2];
  float* hgru_f[2];  unsigned short* hgru_b[2];
  for (int i = 0; i < 2; ++i) {
    hesn_f[i] = (float*)alloc((size_t)64 * 2048 * 4);
    hesn_b[i] = (unsigned short*)alloc((size_t)64 * 2048 * 2);
    hgru_f[i] = (float*)alloc((size_t)64 * 512 * 4);
    hgru_b[i] = (unsigned short*)alloc((size_t)64 * 512 * 2);
  }
  float*          proj_f  = (float*)alloc((size_t)64 * 512 * 4);
  unsigned short* proj_bf = (unsigned short*)alloc((size_t)64 * 512 * 2);
  unsigned short* comb_bf = (unsigned short*)alloc((size_t)64 * 512 * 2);
  unsigned short* hid_bf  = (unsigned short*)alloc((size_t)64 * 512 * 2);

  // ---- prep: pack weights to bf16 WMMA-B layout, convert x, zero h0 ----
  auto pack = [&](const float* src, unsigned short* dst, int K, int N, int ks, int ns) {
    int total = K * N;
    pack_b_kernel<<<(total + 255) / 256, 256, 0, stream>>>(src, dst, K, N, ks, ns);
  };
  pack(W_in,    WinP,  128,  2048, 2048, 1);     // x @ W_in
  pack(W_res,   WresP, 2048, 2048, 2048, 1);     // h @ W_res
  pack(gW_ih,   WihP,  128,  1536, 1,    128);   // x @ W_ih^T
  pack(gW_hh,   WhhP,  512,  1536, 1,    512);   // h @ W_hh^T
  pack(proj_W,  projP, 2048, 512,  1,    2048);  // esn_h @ proj_W^T
  pack(gate_W,  gateP, 1024, 512,  1,    1024);  // gate_in @ gate_W^T
  pack(head_W1, h1P,   512,  512,  1,    512);   // combined @ head_W1^T
  pack(head_W2, h2P,   512,  128,  1,    512);   // hidden @ head_W2^T

  {
    int n = 64 * 512 * 128;
    f32_to_bf16_kernel<<<(n + 255) / 256, 256, 0, stream>>>(x, xbf, n);
  }
  zero_kernel<<<(64 * 2048 + 255) / 256, 256, 0, stream>>>((unsigned int*)hesn_f[0], 64 * 2048);
  zero_kernel<<<(64 * 1024 + 255) / 256, 256, 0, stream>>>((unsigned int*)hesn_b[0], 64 * 1024);
  zero_kernel<<<(64 * 512  + 255) / 256, 256, 0, stream>>>((unsigned int*)hgru_f[0], 64 * 512);
  zero_kernel<<<(64 * 256  + 255) / 256, 256, 0, stream>>>((unsigned int*)hgru_b[0], 64 * 256);

  // ---- 512 fused recurrent steps (ESN + GRU), ping-pong buffers ----
  for (int t = 0; t < 512; ++t) {
    int in = t & 1, outb = in ^ 1;
    step_kernel<<<80, 256, 0, stream>>>(
        xbf, t, WinP, WresP, WihP, WhhP, gb_ih, gb_hh,
        hesn_f[in], hesn_b[in], hesn_f[outb], hesn_b[outb],
        hgru_f[in], hgru_b[in], hgru_f[outb], hgru_b[outb]);
  }
  // after t=511, final states are in buffer 0
  float* esn_fin = hesn_f[0];  unsigned short* esn_fin_b = hesn_b[0];
  float* gru_fin = hgru_f[0];  unsigned short* gru_fin_b = hgru_b[0];
  (void)esn_fin;

  // ---- tail: proj -> gate/mix -> head1(relu) -> head2 ----
  tail_gemm_kernel<<<16, 256, 0, stream>>>(
      esn_fin_b, 2048, esn_fin_b, 2048, 64, projP, 64, 32,
      proj_b, 0, nullptr, nullptr, 0, proj_f, proj_bf, 512);
  tail_gemm_kernel<<<16, 256, 0, stream>>>(
      proj_bf, 512, gru_fin_b, 512, 16, gateP, 32, 32,
      gate_b, 1, proj_f, gru_fin, 512, nullptr, comb_bf, 512);
  tail_gemm_kernel<<<16, 256, 0, stream>>>(
      comb_bf, 512, comb_bf, 512, 16, h1P, 16, 32,
      head_b1, 2, nullptr, nullptr, 0, nullptr, hid_bf, 512);
  tail_gemm_kernel<<<4, 256, 0, stream>>>(
      hid_bf, 512, hid_bf, 512, 16, h2P, 16, 8,
      head_b2, 3, nullptr, nullptr, 0, out, nullptr, 128);
}
